// GPML_LLF_38397007626880
// MI455X (gfx1250) — matrix operand order repaired and checked by reading, not compiled, not run on hardware
//
#include <hip/hip_runtime.h>
#include <hip/hip_bf16.h>
#include <math.h>

typedef __attribute__((ext_vector_type(16))) _Float16 v16h;
typedef __attribute__((ext_vector_type(8)))  _Float16 v8h;
typedef __attribute__((ext_vector_type(8)))  float    v8f;
typedef __attribute__((ext_vector_type(4)))  unsigned int u32x4;
typedef __attribute__((ext_vector_type(8)))  int          i32x8;
typedef __attribute__((ext_vector_type(4)))  int          i32x4;

#define B_ 32
#define S_ 64
#define T_ 256
#define L_ 8
#define D_ 128
#define H_ 128
#define N_ (B_ * S_)   // 2048 (b,s) pairs

#if defined(__gfx1250__) && __has_builtin(__builtin_amdgcn_tensor_load_to_lds) && \
    __has_builtin(__builtin_amdgcn_s_wait_tensorcnt)
#define USE_TDM 1
#else
#define USE_TDM 0
#endif

__device__ __forceinline__ v16h cat8(v8h lo, v8h hi) {
  return __builtin_shufflevector(lo, hi, 0,1,2,3,4,5,6,7,8,9,10,11,12,13,14,15);
}

#if USE_TDM
// ---------------------------------------------------------------------------
// Tensor Data Mover: 2D tile of 2-byte elements, global -> LDS.
// D# packing per CDNA5 ISA (group0) / §8.4 (group1):
//   g0: [1:0]=count=1, [63:32]=lds_addr, [120:64]=global_addr, [127:126]=2
//   g1: [17:16]=data_size(1 -> 2B), [79:48]=tensor_dim0, [111:80]=tensor_dim1,
//       [127:112]=tile_dim0, [143:128]=tile_dim1, [207:160]=tensor_dim0_stride
// Groups 2/3 passed as zeros (2D tile: tile_dim2/3 = 0 = unused).
// This toolchain's builtin is the 6-arg form:
//   (u32x4 g0, i32x8 g1, i32x4 g2, i32x4 g3, i32x8 extra, i32 cpol)
// ---------------------------------------------------------------------------
__device__ __forceinline__ void tdm_load_2d_f16(const void* gptr, void* lptr,
                                                unsigned dim0, unsigned dim1,
                                                unsigned tile0, unsigned tile1,
                                                unsigned long long stride0) {
  unsigned long long ga = (unsigned long long)(size_t)gptr;
  unsigned la = (unsigned)(size_t)lptr;   // generic LDS addr: low 32 bits = LDS offset
  u32x4 g0;
  g0[0] = 1u;                                                 // count=1 (valid)
  g0[1] = la;                                                 // lds_addr
  g0[2] = (unsigned)(ga & 0xffffffffu);                       // global_addr lo
  g0[3] = (unsigned)((ga >> 32) & 0x01ffffffu) | (2u << 30);  // addr hi | type=2
  i32x8 g1;
  g1[0] = (int)(1u << 16);                                    // data_size = 2B
  g1[1] = (int)((dim0 & 0xffffu) << 16);                      // tensor_dim0 lo16
  g1[2] = (int)((dim0 >> 16) | ((dim1 & 0xffffu) << 16));     // dim0 hi | dim1 lo
  g1[3] = (int)((dim1 >> 16) | ((tile0 & 0xffffu) << 16));    // dim1 hi | tile_dim0
  g1[4] = (int)(tile1 & 0xffffu);                             // tile_dim1 (tile_dim2=0)
  g1[5] = (int)(unsigned)(stride0 & 0xffffffffu);             // dim0_stride lo32
  g1[6] = (int)(unsigned)((stride0 >> 32) & 0xffffu);         // dim0_stride hi16
  g1[7] = 0;
  i32x4 z4 = {0, 0, 0, 0};
  i32x8 z8 = {0, 0, 0, 0, 0, 0, 0, 0};
  __builtin_amdgcn_tensor_load_to_lds(g0, g1, z4, z4, z8, 0);
}
#endif

// ---------------------------------------------------------------------------
// Kernel 0: precision prep.  eps f32 -> f16 (same [n][l][t] layout) and
// W2 f32 [k][d] -> W2t f16 [d][k] so GEMM B-operands are contiguous 8-half
// runs.  16.8 MB read + 8.4 MB write: ~1 us at 23.3 TB/s.
// ---------------------------------------------------------------------------
__global__ void k_prep(const float* __restrict__ eps,
                       const float* __restrict__ W2,
                       _Float16* __restrict__ epsh,
                       _Float16* __restrict__ W2t) {
  const size_t i = (size_t)blockIdx.x * 256 + threadIdx.x;
  epsh[i] = (_Float16)eps[i];                 // grid sized exactly N_*L_*T_
  if (i < (size_t)H_ * D_) {
    const int k = (int)(i >> 7), d = (int)(i & 127);
    W2t[d * H_ + k] = (_Float16)W2[i];
  }
}

// ---------------------------------------------------------------------------
// Kernel 1: K[l] = s2*exp(-dt^2/(2 tau_l^2)) + jitter*I, fp32 Cholesky in
// workspace, emit Lc as f16 row-major (upper zeroed) for WMMA A loads.
// ---------------------------------------------------------------------------
__global__ void k_chol(const float* __restrict__ times,
                       const float* __restrict__ log_taus,
                       float* __restrict__ Kbuf,
                       _Float16* __restrict__ Lch) {
  const int l   = blockIdx.x;
  const int tid = threadIdx.x;          // 0..255 == row index
  float* A = Kbuf + (size_t)l * T_ * T_;
  const float tau  = __expf(log_taus[l]);
  const float inv2 = 1.0f / (2.0f * tau * tau);
  const float s2   = (1.0f - 0.001f) * (1.0f - 0.001f);
  const float ti   = times[tid];
  for (int j = 0; j < T_; ++j) {
    float dt = ti - times[j];
    float v  = s2 * __expf(-dt * dt * inv2);
    if (j == tid) v += 0.01f * 0.01f;
    A[tid * T_ + j] = v;
  }
  __syncthreads();
  for (int j = 0; j < T_; ++j) {
    if (tid == j) A[j * T_ + j] = sqrtf(A[j * T_ + j]);
    __syncthreads();
    const float ajj = A[j * T_ + j];
    if (tid > j) A[tid * T_ + j] /= ajj;
    __syncthreads();
    if (tid > j) {
      const float aij = A[tid * T_ + j];
      for (int k = j + 1; k <= tid; ++k)
        A[tid * T_ + k] -= aij * A[k * T_ + j];
    }
    __syncthreads();
  }
  for (int j = 0; j < T_; ++j)
    Lch[(size_t)l * T_ * T_ + (size_t)tid * T_ + j] =
        (j <= tid) ? (_Float16)A[tid * T_ + j] : (_Float16)0.0f;
}

// ---------------------------------------------------------------------------
// Kernel 2: Z_l = Lc_l [256x256] @ Eps_l [256 x 2048]   (f16 WMMA, f32 acc)
// grid (Mtiles=2, Ntiles=128, L=8); 8 waves/block, one 16x16 C tile per wave.
// Whole 16x256 B panel (8 KB) staged into LDS ONCE (TDM if available), then
// 8 barrier-free WMMA K-steps.  Output stored as Z[n][t][l].
// ---------------------------------------------------------------------------
__global__ void k_zgemm(const _Float16* __restrict__ epsh,
                        const _Float16* __restrict__ Lch,
                        float* __restrict__ Z) {
  __shared__ _Float16 sB[16 * T_];      // [n_local][t]  (t = K, contiguous)
  const int l    = blockIdx.z;
  const int n0   = blockIdx.y * 16;
  const int tid  = threadIdx.x;
  const int wave = tid >> 5;
  const int lane = tid & 31;
  const int m    = lane & 15;
  const bool hi  = lane >= 16;
  const int t0   = blockIdx.x * 128 + wave * 16;

#if USE_TDM
  if (__builtin_amdgcn_readfirstlane(tid >> 5) == 0) {   // scalar branch: wave 0 only
    tdm_load_2d_f16(epsh + ((size_t)n0 * L_ + l) * T_, sB,
                    /*dim0=*/T_, /*dim1=*/16, /*tile0=*/T_, /*tile1=*/16,
                    /*stride0=*/(unsigned long long)(L_ * T_));
    __builtin_amdgcn_s_wait_tensorcnt(0);
  }
#else
  for (int i = tid; i < 16 * T_; i += 256) {
    const int nl = i >> 8, tl = i & 255;
    sB[i] = epsh[((size_t)(n0 + nl) * L_ + l) * T_ + tl];
  }
#endif
  __syncthreads();

  const _Float16* Arow = Lch + (size_t)l * T_ * T_ + (size_t)(t0 + m) * T_;
  const int kOffA = hi ? 8 : 0;         // f16 A layout: lanes 0-15 K{0..7,16..23}

  v8f acc;
  #pragma unroll
  for (int i = 0; i < 8; ++i) acc[i] = 0.0f;

  #pragma unroll
  for (int kc = 0; kc < 8; ++kc) {
    const int kb = kc * 32;
    v8h a0 = *(const v8h*)(Arow + kb + kOffA);
    v8h a1 = *(const v8h*)(Arow + kb + kOffA + 16);
    v16h av = cat8(a0, a1);
    const _Float16* bp = &sB[m * T_ + kb + (hi ? 8 : 0)];
    v16h bv = cat8(*(const v8h*)bp, *(const v8h*)(bp + 16));
    acc = __builtin_amdgcn_wmma_f32_16x16x32_f16(false, av, false, bv,
                                                 (short)0, acc, false, false);
  }
  // C layout: lanes 0-15: N=lane, M=r ; lanes 16-31: N=lane-16, M=8+r
  const int nn = n0 + m;
  const int tb = t0 + (hi ? 8 : 0);
  #pragma unroll
  for (int r = 0; r < 8; ++r)
    Z[((size_t)nn * T_ + (tb + r)) * L_ + l] = acc[r];
}

// ---------------------------------------------------------------------------
// Kernel 3: fused decoder + Gaussian logprob.  One block per (b,s).
// W2t (32 KB f16) pulled to LDS by TDM; h=tanh(Z@W1+b1) built directly in the
// WMMA-A register layout; mu=h@W2 via 8x4 v_wmma per 16-row strip; logprob
// epilogue in registers; 16-lane shuffle reduction over D; mu never stored.
// ---------------------------------------------------------------------------
__global__ void k_decode(const float* __restrict__ X,
                         const float* __restrict__ logR,
                         const float* __restrict__ W1,
                         const float* __restrict__ b1,
                         const _Float16* __restrict__ W2t,
                         const float* __restrict__ b2,
                         const float* __restrict__ Z,
                         float* __restrict__ ll) {
  __shared__ float    s_w1[L_ * H_];
  __shared__ float    s_b1[H_], s_b2[D_], s_rinv[D_];
  __shared__ _Float16 s_w2t[D_ * H_];   // [d][k], k contiguous
  __shared__ float    s_part[16];
  __shared__ float    s_slr;

  const int n   = blockIdx.x;
  const int b   = n / S_;
  const int tid = threadIdx.x;

#if USE_TDM
  if (__builtin_amdgcn_readfirstlane(tid >> 5) == 0) {   // wave 0 issues the DMA
    tdm_load_2d_f16(W2t, s_w2t, /*dim0=*/D_ * H_, /*dim1=*/1,
                    /*tile0=*/D_ * H_, /*tile1=*/1,
                    /*stride0=*/(unsigned long long)(D_ * H_));
    __builtin_amdgcn_s_wait_tensorcnt(0);
  }
#else
  for (int i = tid; i < D_ * H_; i += 256) s_w2t[i] = W2t[i];
#endif
  if (tid < D_) {
    s_b1[tid]   = b1[tid];
    s_b2[tid]   = b2[tid];
    s_rinv[tid] = __expf(-logR[tid]);
  }
  for (int i = tid; i < L_ * H_; i += 256) s_w1[i] = W1[i];
  if (tid == 0) {
    float v = 0.0f;
    for (int d = 0; d < D_; ++d) v += logR[d];
    s_slr = v;
  }
  __syncthreads();

  const int wave = tid >> 5;
  const int lane = tid & 31;
  const int m    = lane & 15;
  const bool hi  = lane >= 16;
  const float log2pi   = 1.8378770664093454836f;
  const float log_norm = 0.5f * (s_slr + (float)D_ * log2pi);
  const float* Xb = X + (size_t)b * T_ * D_;

  float wacc = 0.0f;
  for (int it = 0; it < 2; ++it) {
    const int t0 = (wave + it * 8) * 16;
    __builtin_prefetch(Xb + (size_t)t0 * D_, 0, 0);

    float zr[L_];
    const float* zp = Z + ((size_t)n * T_ + (t0 + m)) * L_;
    #pragma unroll
    for (int l = 0; l < L_; ++l) zr[l] = zp[l];

    // h (A operand) directly in f16 WMMA layout: 4 K-chunks of 32
    v16h Am[4];
    #pragma unroll
    for (int c = 0; c < 4; ++c) {
      #pragma unroll
      for (int j = 0; j < 16; ++j) {
        const int k = c * 32 + (hi ? 8 : 0) + (j < 8 ? j : j + 8);
        float a = s_b1[k];
        #pragma unroll
        for (int l = 0; l < L_; ++l) a += zr[l] * s_w1[l * H_ + k];
        Am[c][j] = (_Float16)tanhf(a);
      }
    }

    float q[8];
    #pragma unroll
    for (int r = 0; r < 8; ++r) q[r] = 0.0f;

    #pragma unroll
    for (int nc = 0; nc < 8; ++nc) {
      v8f accv;
      #pragma unroll
      for (int i = 0; i < 8; ++i) accv[i] = 0.0f;
      #pragma unroll
      for (int c = 0; c < 4; ++c) {
        const _Float16* bp = &s_w2t[(nc * 16 + m) * H_ + c * 32 + (hi ? 8 : 0)];
        v16h bv = cat8(*(const v8h*)bp, *(const v8h*)(bp + 16));
        accv = __builtin_amdgcn_wmma_f32_16x16x32_f16(false, Am[c], false, bv,
                                                      (short)0, accv, false, false);
      }
      const int d     = nc * 16 + m;
      const float bb  = s_b2[d];
      const float rin = s_rinv[d];
      #pragma unroll
      for (int r = 0; r < 8; ++r) {
        const int t = t0 + r + (hi ? 8 : 0);
        const float diff = Xb[(size_t)t * D_ + d] - (accv[r] + bb);
        q[r] += diff * diff * rin;
      }
    }

    #pragma unroll
    for (int r = 0; r < 8; ++r) {
      float v = q[r];
      v += __shfl_xor(v, 1, 32);
      v += __shfl_xor(v, 2, 32);
      v += __shfl_xor(v, 4, 32);
      v += __shfl_xor(v, 8, 32);
      if (m == 0) wacc += -0.5f * v - log_norm;
    }
  }
  if (m == 0) s_part[wave * 2 + (hi ? 1 : 0)] = wacc;
  __syncthreads();
  if (tid == 0) {
    float v = 0.0f;
    for (int i = 0; i < 16; ++i) v += s_part[i];
    ll[n] = v;
  }
}

// ---------------------------------------------------------------------------
// Kernel 4: nll[b] = -(logsumexp_s(ll[b,:]) - log S)
// ---------------------------------------------------------------------------
__global__ void k_lse(const float* __restrict__ ll, float* __restrict__ out) {
  __shared__ float sv[64];
  const int b = blockIdx.x, tid = threadIdx.x;
  const float v = ll[b * S_ + tid];
  sv[tid] = v;
  __syncthreads();
  for (int o = 32; o > 0; o >>= 1) {
    if (tid < o) sv[tid] = fmaxf(sv[tid], sv[tid + o]);
    __syncthreads();
  }
  const float mx = sv[0];
  __syncthreads();
  sv[tid] = __expf(v - mx);
  __syncthreads();
  for (int o = 32; o > 0; o >>= 1) {
    if (tid < o) sv[tid] += sv[tid + o];
    __syncthreads();
  }
  if (tid == 0) out[b] = -(mx + logf(sv[0]) - logf((float)S_));
}

// ---------------------------------------------------------------------------
// Launcher.  Workspace layout (bytes):
//   [0,     2 MB)   Kbuf  fp32  8*256*256
//   [2 MB,  3 MB)   Lch   f16   8*256*256
//   [3 MB, 11 MB)   epsh  f16   2048*8*256
//   [11 MB, +32KB)  W2t   f16   128*128
//   [12 MB, 28 MB)  Z     fp32  2048*256*8
//   [28 MB, +8 KB)  ll    fp32  2048
// ---------------------------------------------------------------------------
extern "C" void kernel_launch(void* const* d_in, const int* in_sizes, int n_in,
                              void* d_out, int out_size, void* d_ws, size_t ws_size,
                              hipStream_t stream) {
  const float* X        = (const float*)d_in[0];
  const float* times    = (const float*)d_in[1];
  const float* log_taus = (const float*)d_in[2];
  const float* logR     = (const float*)d_in[3];
  const float* W1       = (const float*)d_in[4];
  const float* b1       = (const float*)d_in[5];
  const float* W2       = (const float*)d_in[6];
  const float* b2       = (const float*)d_in[7];
  const float* eps      = (const float*)d_in[8];
  float* out = (float*)d_out;

  char* ws = (char*)d_ws;
  float*    Kbuf = (float*)(ws);
  _Float16* Lch  = (_Float16*)(ws +  2u * 1024 * 1024);
  _Float16* epsh = (_Float16*)(ws +  3u * 1024 * 1024);
  _Float16* W2t  = (_Float16*)(ws + 11u * 1024 * 1024);
  float*    Z    = (float*)   (ws + 12u * 1024 * 1024);
  float*    ll   = (float*)   (ws + 28u * 1024 * 1024);

  const int prep_blocks = (N_ * L_ * T_) / 256;   // 16384
  k_prep  <<<dim3(prep_blocks),   dim3(256), 0, stream>>>(eps, W2, epsh, W2t);
  k_chol  <<<dim3(L_),            dim3(256), 0, stream>>>(times, log_taus, Kbuf, Lch);
  k_zgemm <<<dim3(2, N_/16, L_),  dim3(256), 0, stream>>>(epsh, Lch, Z);
  k_decode<<<dim3(N_),            dim3(256), 0, stream>>>(X, logR, W1, b1, W2t, b2, Z, ll);
  k_lse   <<<dim3(B_),            dim3(64),  0, stream>>>(ll, out);
}
